// CrossAttentionRope_23295902614139
// MI455X (gfx1250) — matrix operand order repaired and checked by reading, not compiled, not run on hardware
//
#include <hip/hip_runtime.h>

// Problem constants
#define BQ 4
#define NQ 2048
#define MK 2048
#define CH 1024
#define NH 16
#define HD 64

typedef __attribute__((ext_vector_type(16))) _Float16 v16h;
typedef __attribute__((ext_vector_type(8)))  float    v8f;
typedef __attribute__((ext_vector_type(4)))  float    v4f;

__device__ __forceinline__ v8f wmma16(v16h a, v16h b, v8f c) {
  // D = A(16x32 f16) * B(32x16 f16) + C(16x16 f32)
  return __builtin_amdgcn_wmma_f32_16x16x32_f16(
      /*neg_a=*/false, a, /*neg_b=*/false, b,
      /*c_mod=*/(short)0, c, /*reuse_a=*/false, /*reuse_b=*/false);
}

// 16-lane (DPP-row) butterfly reductions, pure VALU: v_mov_b32_dpp + op.
// xor1 = quad_perm(1,0,3,2)=0xB1; xor2 = quad_perm(2,3,0,1)=0x4E;
// then ROW_HALF_MIRROR (0x141) acts as xor4 and ROW_MIRROR (0x140) as xor8
// once the smaller groups are uniform. DPP ctrl must be a literal constant.
template <int CTRL>
__device__ __forceinline__ float dpp_bfly(float x) {
  int t = __builtin_amdgcn_update_dpp(0, __float_as_int(x), CTRL, 0xF, 0xF, true);
  return __int_as_float(t);
}
__device__ __forceinline__ float dpp_max16(float x) {
  x = fmaxf(x, dpp_bfly<0xB1>(x));   // xor 1
  x = fmaxf(x, dpp_bfly<0x4E>(x));   // xor 2
  x = fmaxf(x, dpp_bfly<0x141>(x));  // row_half_mirror -> xor 4
  x = fmaxf(x, dpp_bfly<0x140>(x));  // row_mirror      -> xor 8
  return x;
}
__device__ __forceinline__ float dpp_sum16(float x) {
  x += dpp_bfly<0xB1>(x);
  x += dpp_bfly<0x4E>(x);
  x += dpp_bfly<0x141>(x);
  x += dpp_bfly<0x140>(x);
  return x;
}

// ---------------------------------------------------------------------------
// Projection: Y[row, col] = sum_c X[row,c] * W[col,c]  (torch Linear, no bias)
// f32 inputs converted to f16 on load; f32 WMMA accumulation; f16 output.
// MODE 0: dst[b][h][s][d]   (Q / K layout)
// MODE 1: dst[b][h][d][m]   (V transposed layout for the PV WMMA B-operand)
// grid: (rows/16, C/64), block: 128 (4 waves; wave w owns col tile 4*by+w)
// ---------------------------------------------------------------------------
template <int MODE>
__global__ __launch_bounds__(128) void proj_f32_kernel(
    const float* __restrict__ X, const float* __restrict__ W,
    _Float16* __restrict__ dst) {
  const int lane = threadIdx.x & 31;
  const int wave = threadIdx.x >> 5;
  const int r    = lane & 15;
  const bool hi  = lane >= 16;
  const int kb   = hi ? 16 : 0;
  const int row0 = blockIdx.x * 16;
  const int col0 = (blockIdx.y * 4 + wave) * 16;

  const float* xrow = X + (size_t)(row0 + r) * CH;
  const float* wrow = W + (size_t)(col0 + r) * CH + (hi ? 16 : 0);

  v8f acc = {};
  for (int c0 = 0; c0 < CH; c0 += 32) {
    // A fragment: lane r holds row row0+r, K = {kb..kb+15}
    const v4f* xa = (const v4f*)(xrow + c0 + kb);
    v4f x0 = xa[0], x1 = xa[1];
    const v4f* xb = (const v4f*)(xrow + c0 + kb + 8);
    v4f x2 = xb[0], x3 = xb[1];
    // B fragment: lane holds B^T row (= W row col0+r), 16 contiguous K
    const v4f* wp = (const v4f*)(wrow + c0);
    v4f w0 = wp[0], w1 = wp[1], w2 = wp[2], w3 = wp[3];
    v16h a, b;
#pragma unroll
    for (int i = 0; i < 4; ++i) {
      a[i]      = (_Float16)x0[i];
      a[4 + i]  = (_Float16)x1[i];
      a[8 + i]  = (_Float16)x2[i];
      a[12 + i] = (_Float16)x3[i];
      b[i]      = (_Float16)w0[i];
      b[4 + i]  = (_Float16)w1[i];
      b[8 + i]  = (_Float16)w2[i];
      b[12 + i] = (_Float16)w3[i];
    }
    acc = wmma16(a, b, acc);
  }

  // C layout: acc[i] -> row row0 + i + (hi?8:0), col col0 + (lane&15)
  const int col = col0 + r;
  const int h = col >> 6;      // head
  const int d = col & 63;      // dim within head
#pragma unroll
  for (int i = 0; i < 8; ++i) {
    const int m  = row0 + i + (hi ? 8 : 0);
    const int bb = m >> 11;       // batch  (rows-per-batch = 2048)
    const int s  = m & 2047;      // seq position
    size_t off;
    if (MODE == 0)
      off = ((size_t)(bb * NH + h) * NQ + s) * HD + d;
    else
      off = ((size_t)(bb * NH + h) * HD + d) * (size_t)MK + s;
    dst[off] = (_Float16)acc[i];
  }
}

// ---------------------------------------------------------------------------
// Flash attention: one wave per 16-query tile, 64-key steps, online softmax
// in exp2 domain. Qh,Kh: [B,H,S,64] f16 (Q pre-scaled by SCALE*log2e).
// Vt: [B,H,64,M] f16. Row max reduced once per 64-key step via DPP16
// butterflies (VALU only); denominator kept as per-lane partial, reduced once
// at the end. 16 WMMA per step (8 QK^T + 8 PV) amortize the softmax VALU.
// Output Xa: [B,N,C] f16. grid: (N/16/8, B*H), block: 256 (8 waves).
// ---------------------------------------------------------------------------
__global__ __launch_bounds__(256) void flash_attn_kernel(
    const _Float16* __restrict__ Qh, const _Float16* __restrict__ Kh,
    const _Float16* __restrict__ Vt, _Float16* __restrict__ Xa) {
  __shared__ _Float16 pbuf[8][16 * 64];  // per-wave P staging (C->A relayout)

  const int lane = threadIdx.x & 31;
  const int wave = threadIdx.x >> 5;
  const int r    = lane & 15;
  const bool hi  = lane >= 16;
  const int kb   = hi ? 16 : 0;
  const int bh   = blockIdx.y;                       // b*16 + h
  const int q0   = (blockIdx.x * 8 + wave) * 16;     // query tile base
  const float scale = 0.125f * 1.44269504088896340736f;  // D^-0.5 * log2(e)

  // Load Q fragments for d-chunks [0,32) and [32,64), pre-scaled.
  const _Float16* qrow = Qh + ((size_t)bh * NQ + q0 + r) * HD;
  v16h qa0, qa1;
#pragma unroll
  for (int i = 0; i < 8; ++i) {
    qa0[i]     = (_Float16)((float)qrow[kb + i] * scale);
    qa0[8 + i] = (_Float16)((float)qrow[kb + 8 + i] * scale);
    qa1[i]     = (_Float16)((float)qrow[32 + kb + i] * scale);
    qa1[8 + i] = (_Float16)((float)qrow[32 + kb + 8 + i] * scale);
  }

  v8f o0 = {}, o1 = {}, o2 = {}, o3 = {};    // O: 16 x 64 accumulator
  float mrun[8], lpart[8];
#pragma unroll
  for (int i = 0; i < 8; ++i) { mrun[i] = -1e30f; lpart[i] = 0.0f; }

  _Float16* pb = &pbuf[wave][0];
  const _Float16* kbase = Kh + (size_t)bh * MK * HD;
  const _Float16* vbase = Vt + (size_t)bh * HD * MK;

  for (int kt = 0; kt < MK; kt += 64) {
    // ---- S = Q * K^T for four 16-key subtiles (8 WMMA) -------------------
    const _Float16* k0p = kbase + (size_t)(kt + r) * HD + (hi ? 16 : 0);
    v8f s0 = {}, s1 = {}, s2 = {}, s3 = {};
    {
      v16h b0 = *(const v16h*)(k0p);                 // keys +0..15,  d 0..31
      v16h b1 = *(const v16h*)(k0p + 32);            //               d 32..63
      v16h b2 = *(const v16h*)(k0p + 16 * HD);       // keys +16..31
      v16h b3 = *(const v16h*)(k0p + 16 * HD + 32);
      v16h b4 = *(const v16h*)(k0p + 32 * HD);       // keys +32..47
      v16h b5 = *(const v16h*)(k0p + 32 * HD + 32);
      v16h b6 = *(const v16h*)(k0p + 48 * HD);       // keys +48..63
      v16h b7 = *(const v16h*)(k0p + 48 * HD + 32);
      s0 = wmma16(qa0, b0, s0);  s0 = wmma16(qa1, b1, s0);
      s1 = wmma16(qa0, b2, s1);  s1 = wmma16(qa1, b3, s1);
      s2 = wmma16(qa0, b4, s2);  s2 = wmma16(qa1, b5, s2);
      s3 = wmma16(qa0, b6, s3);  s3 = wmma16(qa1, b7, s3);
    }

    if (kt + 64 < MK) {  // global_prefetch_b8 for next K / V tiles
      __builtin_prefetch((const void*)(k0p + 64 * HD), 0, 1);
      __builtin_prefetch((const void*)(vbase + (size_t)r * MK + kt + 64), 0, 1);
    }

    // ---- online softmax (exp2 domain); one DPP16 row-max per 64 keys -----
#pragma unroll
    for (int i = 0; i < 8; ++i) {
      const float mx =
          dpp_max16(fmaxf(fmaxf(s0[i], s1[i]), fmaxf(s2[i], s3[i])));
      const float nm = fmaxf(mrun[i], mx);
      const float alpha = exp2f(mrun[i] - nm);
      mrun[i] = nm;
      const float p0 = exp2f(s0[i] - nm);
      const float p1 = exp2f(s1[i] - nm);
      const float p2 = exp2f(s2[i] - nm);
      const float p3 = exp2f(s3[i] - nm);
      lpart[i] = lpart[i] * alpha + ((p0 + p1) + (p2 + p3));
      o0[i] *= alpha; o1[i] *= alpha; o2[i] *= alpha; o3[i] *= alpha;
      // stage P (row-major 16x64 f16) in this wave's LDS slab
      _Float16* pw = pb + (i + (hi ? 8 : 0)) * 64 + r;
      pw[0]  = (_Float16)p0;
      pw[16] = (_Float16)p1;
      pw[32] = (_Float16)p2;
      pw[48] = (_Float16)p3;
    }

    // ---- re-read P in A-fragment layout (in-order LDS within the wave) ---
    v16h pa0, pa1;
    const _Float16* pr = pb + r * 64;
#pragma unroll
    for (int i = 0; i < 8; ++i) {
      pa0[i]     = pr[kb + i];          // keys kt..kt+31
      pa0[8 + i] = pr[kb + 8 + i];
      pa1[i]     = pr[32 + kb + i];     // keys kt+32..kt+63
      pa1[8 + i] = pr[32 + kb + 8 + i];
    }

    // ---- O += P * V  (B operand = Vt rows, contiguous keys; 8 WMMA) ------
    const _Float16* vr = vbase + (size_t)r * MK + kt + (hi ? 16 : 0);
    o0 = wmma16(pa0, *(const v16h*)(vr),                o0);
    o1 = wmma16(pa0, *(const v16h*)(vr + 16 * MK),      o1);
    o2 = wmma16(pa0, *(const v16h*)(vr + 32 * MK),      o2);
    o3 = wmma16(pa0, *(const v16h*)(vr + 48 * MK),      o3);
    o0 = wmma16(pa1, *(const v16h*)(vr + 32),           o0);
    o1 = wmma16(pa1, *(const v16h*)(vr + 16 * MK + 32), o1);
    o2 = wmma16(pa1, *(const v16h*)(vr + 32 * MK + 32), o2);
    o3 = wmma16(pa1, *(const v16h*)(vr + 48 * MK + 32), o3);
  }

  // ---- final denominator (one cross-lane reduction), normalize, store ----
  const int b = bh >> 4, h = bh & 15;
#pragma unroll
  for (int i = 0; i < 8; ++i) {
    const float inv = 1.0f / dpp_sum16(lpart[i]);
    const int qi = q0 + i + (hi ? 8 : 0);
    const size_t base = ((size_t)(b * NQ + qi)) * CH + h * HD + r;
    Xa[base]      = (_Float16)(o0[i] * inv);
    Xa[base + 16] = (_Float16)(o1[i] * inv);
    Xa[base + 32] = (_Float16)(o2[i] * inv);
    Xa[base + 48] = (_Float16)(o3[i] * inv);
  }
}

// ---------------------------------------------------------------------------
// Output projection: out[row,col] = sum_c Xa[row,c]*Wp[col,c] + bp[col], f32
// ---------------------------------------------------------------------------
__global__ __launch_bounds__(128) void proj_out_kernel(
    const _Float16* __restrict__ Xh, const float* __restrict__ W,
    const float* __restrict__ bias, float* __restrict__ out) {
  const int lane = threadIdx.x & 31;
  const int wave = threadIdx.x >> 5;
  const int r    = lane & 15;
  const bool hi  = lane >= 16;
  const int kb   = hi ? 16 : 0;
  const int row0 = blockIdx.x * 16;
  const int col0 = (blockIdx.y * 4 + wave) * 16;

  const _Float16* xrow = Xh + (size_t)(row0 + r) * CH;
  const float*    wrow = W + (size_t)(col0 + r) * CH + (hi ? 16 : 0);
  const float     bcol = bias[col0 + r];

  v8f acc = {};
  for (int c0 = 0; c0 < CH; c0 += 32) {
    v16h a, b;
    const _Float16* xr = xrow + c0;
#pragma unroll
    for (int i = 0; i < 8; ++i) {
      a[i]     = xr[kb + i];
      a[8 + i] = xr[kb + 8 + i];
    }
    const v4f* wp = (const v4f*)(wrow + c0);
    v4f w0 = wp[0], w1 = wp[1], w2 = wp[2], w3 = wp[3];
#pragma unroll
    for (int i = 0; i < 4; ++i) {
      b[i]      = (_Float16)w0[i];
      b[4 + i]  = (_Float16)w1[i];
      b[8 + i]  = (_Float16)w2[i];
      b[12 + i] = (_Float16)w3[i];
    }
    acc = wmma16(a, b, acc);
  }

  const int col = col0 + r;
#pragma unroll
  for (int i = 0; i < 8; ++i) {
    const int m = row0 + i + (hi ? 8 : 0);
    out[(size_t)m * CH + col] = acc[i] + bcol;
  }
}

// ---------------------------------------------------------------------------
extern "C" void kernel_launch(void* const* d_in, const int* in_sizes, int n_in,
                              void* d_out, int out_size, void* d_ws,
                              size_t ws_size, hipStream_t stream) {
  const float* query = (const float*)d_in[0];
  const float* key_  = (const float*)d_in[1];
  const float* value = (const float*)d_in[2];
  const float* Wq    = (const float*)d_in[3];
  const float* Wk    = (const float*)d_in[4];
  const float* Wv    = (const float*)d_in[5];
  const float* Wp    = (const float*)d_in[6];
  const float* bp    = (const float*)d_in[7];
  float* out = (float*)d_out;

  const size_t elems = (size_t)BQ * NH * NQ * HD;  // 8,388,608 halves each
  _Float16* Qh = (_Float16*)d_ws;
  _Float16* Kh = Qh + elems;
  _Float16* Vt = Kh + elems;
  _Float16* Xa = Vt + elems;   // [B,N,C] f16

  dim3 pgrid((BQ * NQ) / 16, CH / 64);   // (512, 16)
  dim3 pblk(128);
  proj_f32_kernel<0><<<pgrid, pblk, 0, stream>>>(query, Wq, Qh);
  proj_f32_kernel<0><<<pgrid, pblk, 0, stream>>>(key_,  Wk, Kh);
  proj_f32_kernel<1><<<pgrid, pblk, 0, stream>>>(value, Wv, Vt);

  dim3 agrid((NQ / 16) / 8, BQ * NH);    // (16, 64)
  dim3 ablk(256);
  flash_attn_kernel<<<agrid, ablk, 0, stream>>>(Qh, Kh, Vt, Xa);

  proj_out_kernel<<<pgrid, pblk, 0, stream>>>(Xa, Wp, bp, out);
}
